// MessagePassingLayer_7129645712025
// MI455X (gfx1250) — matrix operand order repaired and checked by reading, not compiled, not run on hardware
//
#include <hip/hip_runtime.h>
#include <hip/hip_bf16.h>

// ---------------------------------------------------------------------------
// MessagePassingLayer on MI455X (gfx1250), wave32 + WMMA f16 + async-to-LDS.
//
// Factorizations:
//   P = h@W1a + b1, Q = h@W1b  (per node, not per edge)
//   hidden_ij = relu([E_ij | I16] @ [W1c ; Q_jtile] + P_i)   <- one K=32 WMMA
//   aggregated_i = (sum_j adj_ij * hidden_ij) @ W2 + (sum_j adj_ij) * b2
// All weights pre-swizzled to f16 B-fragment layout so fragment loads are
// two global_load_b128 per wave instead of 16 scalar b32 + cvt chains.
// Edge tiles are double-buffered into LDS via global_load_async_to_lds_b128.
// ---------------------------------------------------------------------------

typedef __attribute__((ext_vector_type(16))) _Float16 v16h;
typedef __attribute__((ext_vector_type(8)))  float    v8f;

#define NODE_DIM 128
#define EDGE_DIM 16
#define MSG_DIM  128
#define HID      256
#define BATCH    8
#define NNODE    256
#define ROWS     (BATCH * NNODE)   // 2048 flattened nodes

__device__ __forceinline__ v8f wmma16(v16h a, v16h b, v8f c) {
  return __builtin_amdgcn_wmma_f32_16x16x32_f16(false, a, false, b, (short)0, c,
                                                false, false);
}

// A fragment (16x32 f16) from row-major f16 (LDS), leading dim ld.
// lanes 0-15: M=lane, halves 0-7 = K0-7, halves 8-15 = K16-23;
// lanes 16-31: M=lane-16, halves = K8-15 / K24-31.
__device__ __forceinline__ v16h load_a_f16(const _Float16* src, int ld, int lane) {
  const int m  = lane & 15;
  const int kb = (lane >> 4) << 3;
  v16h a;
#pragma unroll
  for (int t = 0; t < 8; ++t) {
    a[t]     = src[m * ld + kb + t];
    a[8 + t] = src[m * ld + 16 + kb + t];
  }
  return a;
}

// B fragment from pre-swizzled f16 fragment array: one 32B vector load.
// Layout: frag[((kb * nct + ct) * 32 + lane)*16 + t]
__device__ __forceinline__ v16h load_b_pre(const _Float16* frag, int nct, int kb,
                                           int ct, int lane) {
  const v16h* p = (const v16h*)frag;
  return p[((size_t)kb * nct + ct) * 32 + lane];
}

// Async global -> LDS copy of 16 bytes (gfx1250 ASYNCcnt path).
__device__ __forceinline__ void async_copy_b128(unsigned int lds_off,
                                                const float* gsrc) {
  asm volatile("global_load_async_to_lds_b128 %0, %1, off"
               :: "v"(lds_off), "v"(gsrc)
               : "memory");
}
__device__ __forceinline__ void wait_asynccnt0() {
  asm volatile("s_wait_asynccnt 0x0" ::: "memory");
}

// ---------------------------------------------------------------------------
// Prep: convert a row-major f32 weight [Kreal x Ncols] into f16 B-fragment
// order, zero-padding K up to Kpad (multiple of 32).
// ---------------------------------------------------------------------------
__global__ __launch_bounds__(256)
void prep_weight_frag(const float* __restrict__ W, _Float16* __restrict__ F,
                      int Kreal, int Kpad, int Ncols) {
  const int idx   = blockIdx.x * 256 + threadIdx.x;
  const int total = Kpad * Ncols;
  if (idx >= total) return;
  const int t    = idx & 15;
  const int lane = (idx >> 4) & 31;
  const int ctkb = idx >> 9;          // kb * nct + ct
  const int nct  = Ncols >> 4;
  const int kb   = ctkb / nct;
  const int ct   = ctkb - kb * nct;
  const int n    = lane & 15;
  const int kofs = (lane >> 4) << 3;
  const int k    = kb * 32 + kofs + (t & 7) + ((t >> 3) << 4);
  const int col  = ct * 16 + n;
  F[idx] = (k < Kreal) ? (_Float16)W[(size_t)k * Ncols + col] : (_Float16)0.0f;
}

// ---------------------------------------------------------------------------
// Prep: combined [W1c ; Q_jtile] fragments (32x16 each), per (batch, jtile,
// coltile). K rows 0-15 = W1c (edge weights), rows 16-31 = Q rows of jtile.
// QF[((b*256 + jt*16 + ct)*32 + lane)*16 + t]
// ---------------------------------------------------------------------------
__global__ __launch_bounds__(256)
void prep_qf(const float* __restrict__ W1, const float* __restrict__ Q,
             _Float16* __restrict__ QF) {
  const int idx  = blockIdx.x * 256 + threadIdx.x;   // < 8*16*16*512
  const int t    = idx & 15;
  const int lane = (idx >> 4) & 31;
  const int rest = idx >> 9;          // b*256 + jt*16 + ct
  const int ct   = rest & 15;
  const int jt   = (rest >> 4) & 15;
  const int b    = rest >> 8;
  const int n    = lane & 15;
  const int kofs = (lane >> 4) << 3;
  const int k    = kofs + (t & 7) + ((t >> 3) << 4);  // 0..31
  const int col  = ct * 16 + n;
  float v;
  if (k < EDGE_DIM)
    v = W1[(size_t)(2 * NODE_DIM + k) * HID + col];             // W1c
  else
    v = Q[((size_t)b * NNODE + jt * 16 + (k - 16)) * HID + col]; // Q rows
  QF[idx] = (_Float16)v;
}

// ---------------------------------------------------------------------------
// Kernel 1: P = h @ W1a + b1 ; Q = h @ W1b   (fragment weights, f16)
// ---------------------------------------------------------------------------
__global__ __launch_bounds__(512)
void gnn_pq_kernel(const float* __restrict__ h, const _Float16* __restrict__ W1aF,
                   const _Float16* __restrict__ W1bF, const float* __restrict__ b1,
                   float* __restrict__ P, float* __restrict__ Q) {
  __shared__ _Float16 hs[16 * NODE_DIM];
  const int tid  = threadIdx.x;
  const int lane = tid & 31;
  const int wave = tid >> 5;
  const int row0 = blockIdx.x * 16;

  for (int t = tid; t < 16 * NODE_DIM; t += 512)
    hs[t] = (_Float16)h[(size_t)row0 * NODE_DIM + t];
  __syncthreads();

  v8f accp = {};
  v8f accq = {};
#pragma unroll
  for (int ks = 0; ks < NODE_DIM / 32; ++ks) {
    v16h a  = load_a_f16(hs + ks * 32, NODE_DIM, lane);
    v16h bp = load_b_pre(W1aF, HID / 16, ks, wave, lane);
    v16h bq = load_b_pre(W1bF, HID / 16, ks, wave, lane);
    accp = wmma16(a, bp, accp);
    accq = wmma16(a, bq, accq);
  }

  const int   cw   = wave * 16;
  const int   n    = lane & 15;
  const int   mofs = (lane >> 4) * 8;
  const float bias = b1[cw + n];
#pragma unroll
  for (int r = 0; r < 8; ++r) {
    const int row = row0 + mofs + r;
    P[(size_t)row * HID + cw + n] = accp[r] + bias;  // b1 folded into P
    Q[(size_t)row * HID + cw + n] = accq[r];
  }
}

// ---------------------------------------------------------------------------
// Kernel 2: per (b,i) node; 16 waves = 16 hidden col-tiles. Edge 16x16 tiles
// double-buffered into LDS by wave 0 via async-to-LDS; one K=32 WMMA per
// tile per wave computes E@W1c + Q_jtile in a single shot (identity trick).
// ---------------------------------------------------------------------------
__global__ __launch_bounds__(512)
void gnn_edge_kernel(const float* __restrict__ edge, const float* __restrict__ adj,
                     const _Float16* __restrict__ QF, const float* __restrict__ P,
                     float* __restrict__ AGG, float* __restrict__ RS) {
  __shared__ float adjs[NNODE];
  __shared__ float ebuf[2][16 * 16];
  const int tid  = threadIdx.x;
  const int lane = tid & 31;
  const int wave = tid >> 5;
  const int bi   = blockIdx.x;        // flattened b*N + i
  const int b    = bi >> 8;           // N = 256

  const float* adj_row = adj + (size_t)bi * NNODE;
  const float* erow    = edge + (size_t)bi * NNODE * EDGE_DIM;

  if (wave == 0) {
    // Stage edge tile 0: 32 lanes x 32B = 1KB.
    const float* g = erow + lane * 8;
    unsigned int l0 = (unsigned int)(uintptr_t)&ebuf[0][lane * 8];
    async_copy_b128(l0, g);
    async_copy_b128(l0 + 16, g + 4);
    wait_asynccnt0();
  } else if (tid >= 32 && tid < 32 + NNODE) {
    adjs[tid - 32] = adj_row[tid - 32];
  }
  __syncthreads();

  if (tid == 0) {
    float s = 0.f;
    for (int j = 0; j < NNODE; ++j) s += adjs[j];
    RS[bi] = s;  // rowsum for aggregated b2 bias
  }

  const int cw   = wave * 16;
  const int n    = lane & 15;
  const int m    = lane & 15;
  const int kb0  = (lane >> 4) << 3;
  const int mofs = (lane >> 4) * 8;

  // Identity half of A (K 16..31): A[m][16+k'] = (k' == m).
  _Float16 ident[8];
#pragma unroll
  for (int t = 0; t < 8; ++t)
    ident[t] = (_Float16)(((kb0 + t) == m) ? 1.0f : 0.0f);

  const float pv = P[(size_t)bi * HID + cw + n];  // includes b1
  const v16h* qf = (const v16h*)QF + (size_t)b * NNODE * 32;
  float aggv = 0.f;

  for (int jt = 0; jt < NNODE / 16; ++jt) {
    const int buf = jt & 1;
    if (wave == 0 && jt + 1 < NNODE / 16) {
      const float* g = erow + (size_t)(jt + 1) * 256 + lane * 8;
      unsigned int l0 = (unsigned int)(uintptr_t)&ebuf[buf ^ 1][lane * 8];
      async_copy_b128(l0, g);
      async_copy_b128(l0 + 16, g + 4);
    }

    // A = [E_tile | I16]; E half from LDS.
    const float* eb = &ebuf[buf][m * 16 + kb0];
    v16h a;
#pragma unroll
    for (int t = 0; t < 8; ++t) {
      a[t]     = (_Float16)eb[t];
      a[8 + t] = ident[t];
    }
    // B = [W1c ; Q_jtile] fragment: single 32B vector load.
    v16h bq = qf[((size_t)jt * 16 + wave) * 32 + lane];

    v8f acc = {};
    acc = wmma16(a, bq, acc);  // acc = E@W1c + Q_j

#pragma unroll
    for (int r = 0; r < 8; ++r) {
      const int j   = jt * 16 + mofs + r;
      float     hid = fmaxf(acc[r] + pv, 0.f);
      aggv          = fmaf(adjs[j], hid, aggv);
    }

    if (wave == 0 && jt + 1 < NNODE / 16) wait_asynccnt0();
    __syncthreads();
  }

  aggv += __shfl_xor(aggv, 16, 32);
  if (lane < 16) AGG[(size_t)bi * HID + cw + lane] = aggv;
}

// ---------------------------------------------------------------------------
// Kernel 3: per 16-node tile:
//   msg = AGG @ W2 + rowsum*b2 ; uin = [h, msg] (f16 LDS)
//   uh  = relu(uin @ U1 + c1)  ; out = uh @ U2 + c2
// ---------------------------------------------------------------------------
__global__ __launch_bounds__(512)
void gnn_update_kernel(const float* __restrict__ h, const float* __restrict__ AGG,
                       const float* __restrict__ RS, const _Float16* __restrict__ W2F,
                       const float* __restrict__ b2, const _Float16* __restrict__ U1F,
                       const float* __restrict__ c1, const _Float16* __restrict__ U2F,
                       const float* __restrict__ c2, float* __restrict__ out) {
  __shared__ _Float16 Uin[16 * (NODE_DIM + MSG_DIM)];  // 16 x 256
  __shared__ _Float16 Uh[16 * HID];                    // 16 x 256
  __shared__ _Float16 Ua[16 * HID];                    // staged AGG (f16)
  const int tid  = threadIdx.x;
  const int lane = tid & 31;
  const int wave = tid >> 5;
  const int row0 = blockIdx.x * 16;
  const int n    = lane & 15;
  const int mofs = (lane >> 4) * 8;

  // Stage h into Uin[:, 0:128] and AGG into Ua (all threads).
  for (int t = tid; t < 16 * NODE_DIM; t += 512) {
    const int r = t >> 7, c = t & 127;
    Uin[r * 256 + c] = (_Float16)h[(size_t)(row0 + r) * NODE_DIM + c];
  }
  for (int t = tid; t < 16 * HID; t += 512)
    Ua[t] = (_Float16)AGG[(size_t)row0 * HID + t];
  __syncthreads();

  // Waves 0..7: msg = AGG @ W2 + rowsum*b2 -> Uin[:, 128:256].
  if (wave < 8) {
    const int cw  = wave * 16;
    v8f       acc = {};
#pragma unroll
    for (int ks = 0; ks < HID / 32; ++ks) {
      v16h a = load_a_f16(Ua + ks * 32, HID, lane);
      v16h b = load_b_pre(W2F, MSG_DIM / 16, ks, wave, lane);
      acc = wmma16(a, b, acc);
    }
    const float b2v = b2[cw + n];
#pragma unroll
    for (int r = 0; r < 8; ++r) {
      const int m = mofs + r;
      Uin[m * 256 + NODE_DIM + cw + n] =
          (_Float16)(acc[r] + RS[row0 + m] * b2v);
    }
  }
  __syncthreads();

  // u_hidden = relu(Uin @ U1 + c1) (all 16 waves).
  {
    const int cw  = wave * 16;
    v8f       acc = {};
#pragma unroll
    for (int ks = 0; ks < (NODE_DIM + MSG_DIM) / 32; ++ks) {
      v16h a = load_a_f16(Uin + ks * 32, 256, lane);
      v16h b = load_b_pre(U1F, HID / 16, ks, wave, lane);
      acc = wmma16(a, b, acc);
    }
    const float c1v = c1[cw + n];
#pragma unroll
    for (int r = 0; r < 8; ++r) {
      const int m = mofs + r;
      Uh[m * HID + cw + n] = (_Float16)fmaxf(acc[r] + c1v, 0.f);
    }
  }
  __syncthreads();

  // out = Uh @ U2 + c2 (waves 0..7).
  if (wave < 8) {
    const int cw  = wave * 16;
    v8f       acc = {};
#pragma unroll
    for (int ks = 0; ks < HID / 32; ++ks) {
      v16h a = load_a_f16(Uh + ks * 32, HID, lane);
      v16h b = load_b_pre(U2F, NODE_DIM / 16, ks, wave, lane);
      acc = wmma16(a, b, acc);
    }
    const float c2v = c2[cw + n];
#pragma unroll
    for (int r = 0; r < 8; ++r) {
      const int m = mofs + r;
      out[(size_t)(row0 + m) * NODE_DIM + cw + n] = acc[r] + c2v;
    }
  }
}

// ---------------------------------------------------------------------------
extern "C" void kernel_launch(void* const* d_in, const int* in_sizes, int n_in,
                              void* d_out, int out_size, void* d_ws, size_t ws_size,
                              hipStream_t stream) {
  const float* h    = (const float*)d_in[0];
  const float* adj  = (const float*)d_in[1];
  const float* edge = (const float*)d_in[2];
  const float* W1   = (const float*)d_in[3];
  const float* b1   = (const float*)d_in[4];
  const float* W2   = (const float*)d_in[5];
  const float* b2   = (const float*)d_in[6];
  const float* U1   = (const float*)d_in[7];
  const float* c1   = (const float*)d_in[8];
  const float* U2   = (const float*)d_in[9];
  const float* c2   = (const float*)d_in[10];

  // Workspace layout (f32 then f16 fragments; ~8.8 MB total).
  float* P   = (float*)d_ws;                 // 2048*256
  float* Q   = P + (size_t)ROWS * HID;       // 2048*256
  float* AGG = Q + (size_t)ROWS * HID;       // 2048*256
  float* RS  = AGG + (size_t)ROWS * HID;     // 2048
  _Float16* FR   = (_Float16*)(RS + ROWS);   // 32B-aligned fragment region
  _Float16* W1aF = FR;                       // 128*256
  _Float16* W1bF = W1aF + 128 * 256;         // 128*256
  _Float16* W2F  = W1bF + 128 * 256;         // 256*128
  _Float16* U1F  = W2F + 256 * 128;          // 256*256
  _Float16* U2F  = U1F + 256 * 256;          // 256*128
  _Float16* QF   = U2F + 256 * 128;          // 8*16*16*512 = 1,048,576

  prep_weight_frag<<<(128 * 256) / 256, 256, 0, stream>>>(W1, W1aF, 128, 128, 256);
  prep_weight_frag<<<(128 * 256) / 256, 256, 0, stream>>>(W1 + 128 * 256, W1bF,
                                                          128, 128, 256);
  prep_weight_frag<<<(256 * 128) / 256, 256, 0, stream>>>(W2, W2F, 256, 256, 128);
  prep_weight_frag<<<(256 * 256) / 256, 256, 0, stream>>>(U1, U1F, 256, 256, 256);
  prep_weight_frag<<<(256 * 128) / 256, 256, 0, stream>>>(U2, U2F, 256, 256, 128);

  gnn_pq_kernel<<<ROWS / 16, 512, 0, stream>>>(h, W1aF, W1bF, b1, P, Q);
  prep_qf<<<(BATCH * 16 * 16 * 512) / 256, 256, 0, stream>>>(W1, Q, QF);
  gnn_edge_kernel<<<ROWS, 512, 0, stream>>>(edge, adj, QF, P, AGG, RS);
  gnn_update_kernel<<<ROWS / 16, 512, 0, stream>>>(h, AGG, RS, W2F, b2, U1F, c1,
                                                   U2F, c2, (float*)d_out);
}
